// PairEncoderLayer_9723805958663
// MI455X (gfx1250) — matrix operand order repaired and checked by reading, compiled, not run on hardware
//
#include <hip/hip_runtime.h>

typedef __attribute__((ext_vector_type(16))) __bf16 v16bf;
typedef __attribute__((ext_vector_type(8)))  __bf16 v8bf;
typedef __attribute__((ext_vector_type(8)))  float  v8f;
typedef __attribute__((ext_vector_type(4)))  unsigned int u32x4;
typedef __attribute__((ext_vector_type(8)))  int i32x8;
typedef __attribute__((ext_vector_type(4)))  int i32x4;

#define NB     8
#define SEQ    1024
#define DMODEL 256
#define NH     8
#define HD     64      // KD == VD
#define EDIM   512     // NH*HD
#define DFF    2048

#define HAVE_TDM (__has_builtin(__builtin_amdgcn_tensor_load_to_lds) && \
                  __has_builtin(__builtin_amdgcn_s_wait_tensorcnt))

// ---------- WMMA fragment helpers (bf16 16x16x32, f32 accum) ----------

// A fragment: 16x32 (MxK), source row-major with leading dim ld, K window [k0,k0+32)
// Layout (ISA 7.12.2): lanes 0-15 -> rows, K 0..7 & 16..23 ; lanes 16-31 -> rows, K 8..15 & 24..31
__device__ __forceinline__ v16bf load_a_frag(const __bf16* base, int ld, int k0) {
  int lane = threadIdx.x & 31;
  int r  = lane & 15;
  int kb = (lane >> 4) << 3;                  // 0 or 8
  const __bf16* p = base + (size_t)r * ld + k0 + kb;
  v8bf lo = *(const v8bf*)(p);                // K = kb + 0..7
  v8bf hi = *(const v8bf*)(p + 16);           // K = kb + 16..23
  v16bf a;
#pragma unroll
  for (int i = 0; i < 8; ++i) { a[i] = lo[i]; a[i + 8] = hi[i]; }
  return a;
}

// B fragment: 32x16 (KxN). Source stored TRANSPOSED: srcT[n][k] row-major, stride ld.
// Layout: lanes 0-15 -> col n, K 0..7 & 16..23 ; lanes 16-31 -> col n, K 8..15 & 24..31
__device__ __forceinline__ v16bf load_b_frag(const __bf16* srcT, int ld, int n0, int k0) {
  int lane = threadIdx.x & 31;
  int n  = lane & 15;
  int kb = (lane >> 4) << 3;                  // 0 or 8
  const __bf16* p = srcT + (size_t)(n0 + n) * ld + k0 + kb;
  v8bf lo = *(const v8bf*)(p);
  v8bf hi = *(const v8bf*)(p + 16);
  v16bf b;
#pragma unroll
  for (int i = 0; i < 8; ++i) { b[i] = lo[i]; b[i + 8] = hi[i]; }
  return b;
}

__device__ __forceinline__ v8f wmma_bf16(v16bf a, v16bf b, v8f c) {
  return __builtin_amdgcn_wmma_f32_16x16x32_bf16(false, a, false, b, (short)0, c, false, false);
}

__device__ __forceinline__ v8f zero8() {
  v8f z = {0.f, 0.f, 0.f, 0.f, 0.f, 0.f, 0.f, 0.f};
  return z;
}

// ---------- Weight prep ----------

// Collapse repeat_interleave: Wq'[i,e] = sum_j Wq[16i+j, e]  (16x512); fold 1/8 into Q.
__global__ void prep_wqkv(const float* __restrict__ Wq, const float* __restrict__ Wk,
                          const float* __restrict__ Wv,
                          float* __restrict__ Wqp, float* __restrict__ Wkp,
                          float* __restrict__ Wvp) {
  int gid = blockIdx.x * blockDim.x + threadIdx.x;
  if (gid >= 3 * 16 * EDIM) return;
  int which = gid / (16 * EDIM);
  int r = gid - which * 16 * EDIM;
  int i = r / EDIM;
  int e = r - i * EDIM;
  const float* src = (which == 0) ? Wq : (which == 1) ? Wk : Wv;
  float s = 0.f;
#pragma unroll
  for (int j = 0; j < 16; ++j) s += src[(size_t)(i * 16 + j) * EDIM + e];
  if (which == 0) s *= 0.125f;               // 1/sqrt(KD)
  float* dst = (which == 0) ? Wqp : (which == 1) ? Wkp : Wvp;
  dst[r] = s;
}

// dst[n*K + k] = bf16(src[k*N + n])  (transpose + narrow)
__global__ void transpose_bf16(const float* __restrict__ src, __bf16* __restrict__ dst,
                               int K, int N) {
  int gid = blockIdx.x * blockDim.x + threadIdx.x;
  if (gid >= K * N) return;
  int n = gid / K;
  int k = gid - n * K;
  dst[gid] = (__bf16)src[(size_t)k * N + n];
}

// ---------- QKV projection (K-dim is only 16 after collapse) ----------
__global__ void qkv_proj(const float* __restrict__ plan, const float* __restrict__ knobs,
                         const float* __restrict__ Wqp, const float* __restrict__ Wkp,
                         const float* __restrict__ Wvp,
                         __bf16* __restrict__ Qb, __bf16* __restrict__ Kb,
                         __bf16* __restrict__ Vt) {
  __shared__ float pl[16], kn[16];
  int b = blockIdx.x >> 10;
  int pos = blockIdx.x & 1023;
  if (threadIdx.x < 16)      pl[threadIdx.x]      = plan[(size_t)blockIdx.x * 16 + threadIdx.x];
  else if (threadIdx.x < 32) kn[threadIdx.x - 16] = knobs[(size_t)blockIdx.x * 16 + threadIdx.x - 16];
  __syncthreads();
  for (int e = threadIdx.x; e < EDIM; e += blockDim.x) {
    float q = 0.f, k = 0.f, v = 0.f;
#pragma unroll
    for (int i = 0; i < 16; ++i) {
      q += pl[i] * Wqp[i * EDIM + e];
      k += kn[i] * Wkp[i * EDIM + e];
      v += kn[i] * Wvp[i * EDIM + e];
    }
    int h = e >> 6, c = e & 63;
    size_t qk = ((size_t)(b * NH + h) * SEQ + pos) * HD + c;
    Qb[qk] = (__bf16)q;
    Kb[qk] = (__bf16)k;
    Vt[((size_t)(b * NH + h) * HD + c) * SEQ + pos] = (__bf16)v;
  }
}

// ---------- Fused attention: scores -> softmax over HEADS -> P@V ----------
// One workgroup = 16 query rows of one batch; 8 waves = 8 heads.
// The 256 MB mask is the bandwidth floor: streamed once, non-temporal, with
// per-thread register preload overlapped under the score WMMAs + prefetch of
// the next tile's cache lines.
__global__ void __launch_bounds__(256) attn_kernel(
    const __bf16* __restrict__ Qb, const __bf16* __restrict__ Kb,
    const __bf16* __restrict__ Vt, const float* __restrict__ Wmask,
    __bf16* __restrict__ attn_out) {
  __shared__ float scf[NH * 16 * 32];                 // raw scores per head (16 KB)
  __shared__ __align__(16) __bf16 pbf[NH * 16 * 32];  // head-softmaxed probs (8 KB)

  int b    = blockIdx.x >> 6;
  int s0   = (blockIdx.x & 63) << 4;
  int tid  = threadIdx.x;
  int w    = tid >> 5;                                // wave id == head
  int lane = tid & 31;
  int hg   = b * NH + w;

  const __bf16* Qt = Qb + ((size_t)hg * SEQ + s0) * HD;
  const __bf16* Kh = Kb + (size_t)hg * SEQ * HD;
  const __bf16* Vh = Vt + (size_t)hg * HD * SEQ;

  // per-thread mask pointers for the softmax positions this thread owns
  int M0 = tid >> 5;                                  // 0..7
  int N0 = tid & 31;
  const float* wp0 = Wmask + ((size_t)(b * NH) * SEQ + s0 + M0) * SEQ + N0;
  const float* wp1 = Wmask + ((size_t)(b * NH) * SEQ + s0 + M0 + 8) * SEQ + N0;

  v16bf qa0 = load_a_frag(Qt, HD, 0);
  v16bf qa1 = load_a_frag(Qt, HD, 32);

  v8f O[4];
#pragma unroll
  for (int i = 0; i < 4; ++i) O[i] = zero8();

  for (int t0 = 0; t0 < SEQ; t0 += 32) {
    // --- preload this tile's mask into registers (hidden under score WMMAs);
    //     non-temporal: the 256 MB stream must not evict L2-resident K/V.
    float wreg[16];
#pragma unroll
    for (int hh = 0; hh < NH; ++hh) {
      wreg[hh]     = __builtin_nontemporal_load(wp0 + (size_t)hh * SEQ * SEQ + t0);
      wreg[8 + hh] = __builtin_nontemporal_load(wp1 + (size_t)hh * SEQ * SEQ + t0);
    }
    // --- prefetch next tile's mask lines (128 x 128B lines per tile)
    if (t0 + 32 < SEQ && tid < 128) {
      int hh = tid >> 4, Mr = tid & 15;
      __builtin_prefetch(Wmask + ((size_t)(b * NH + hh) * SEQ + s0 + Mr) * SEQ + t0 + 32, 0, 0);
    }

    // --- scores for this head: [16 s x 32 t]
#pragma unroll
    for (int tc = 0; tc < 2; ++tc) {
      v8f sc = zero8();
      sc = wmma_bf16(qa0, load_b_frag(Kh, HD, t0 + tc * 16, 0),  sc);
      sc = wmma_bf16(qa1, load_b_frag(Kh, HD, t0 + tc * 16, 32), sc);
#pragma unroll
      for (int r = 0; r < 8; ++r) {
        int M = r + ((lane >> 4) << 3);
        int N = tc * 16 + (lane & 15);
        scf[(w * 16 + M) * 32 + N] = sc[r];
      }
    }
    __syncthreads();

    // --- softmax across the 8 heads, per (s,t) element (mask added here)
#pragma unroll
    for (int q2 = 0; q2 < 2; ++q2) {
      int p = tid + q2 * 256;
      float a[NH];
#pragma unroll
      for (int hh = 0; hh < NH; ++hh) a[hh] = scf[hh * 512 + p] + wreg[q2 * 8 + hh];
      float m = a[0];
#pragma unroll
      for (int hh = 1; hh < NH; ++hh) m = fmaxf(m, a[hh]);
      float e[NH]; float sum = 0.f;
#pragma unroll
      for (int hh = 0; hh < NH; ++hh) { e[hh] = __expf(a[hh] - m); sum += e[hh]; }
      float inv = 1.f / sum;
#pragma unroll
      for (int hh = 0; hh < NH; ++hh) pbf[hh * 512 + p] = (__bf16)(e[hh] * inv);
    }
    __syncthreads();

    // --- O += P[16x32] @ V[32x64]
    v16bf pa = load_a_frag(pbf + w * 512, 32, 0);
#pragma unroll
    for (int vc = 0; vc < 4; ++vc)
      O[vc] = wmma_bf16(pa, load_b_frag(Vh, SEQ, vc * 16, t0), O[vc]);
  }

  __bf16* Ao = attn_out + ((size_t)b * SEQ + s0) * EDIM + w * HD;
#pragma unroll
  for (int vc = 0; vc < 4; ++vc)
#pragma unroll
    for (int r = 0; r < 8; ++r) {
      int M = r + ((lane >> 4) << 3);
      int N = vc * 16 + (lane & 15);
      Ao[(size_t)M * EDIM + N] = (__bf16)O[vc][r];
    }
}

// ---------- Fused out-proj + FFN + residual + LayerNorm ----------
__global__ void __launch_bounds__(256) ffn_kernel(
    const __bf16* __restrict__ attn_out,
    const __bf16* __restrict__ WoT, const float* __restrict__ bo,
    const __bf16* __restrict__ W1T, const float* __restrict__ b1,
    const __bf16* __restrict__ W2T, const float* __restrict__ b2,
    const float* __restrict__ gamma, const float* __restrict__ beta,
    float* __restrict__ out) {
  __shared__ __align__(16) __bf16 xa[16 * EDIM];       // 16 KB input tile
  __shared__ float out2f[16 * DMODEL];                 // 16 KB
  __shared__ __align__(16) __bf16 out2b[16 * DMODEL];  // 8 KB
  __shared__ __align__(16) __bf16 hb[16 * DFF];        // 64 KB
  __shared__ float yf[16 * DMODEL];                    // 16 KB
  __shared__ float red1[256], red2[256];
  __shared__ float muA[16], rsA[16];

  int b    = blockIdx.x >> 6;
  int s0   = (blockIdx.x & 63) << 4;
  int tid  = threadIdx.x;
  int w    = tid >> 5;
  int lane = tid & 31;

  const __bf16* Ai = attn_out + ((size_t)b * SEQ + s0) * EDIM;  // 16 KB contiguous

#if HAVE_TDM
  // Tensor Data Mover: one wave DMAs the whole 16 KB tile into LDS.
  if (tid < 32) {
    unsigned long long ga = (unsigned long long)(uintptr_t)Ai;
    unsigned int ldsb = (unsigned int)(uintptr_t)(&xa[0]);  // low 32 bits = LDS byte addr
    // D# group0: count=1, lds_addr, global_addr[56:0], type=2 ("image")
    u32x4 g0 = { 1u, ldsb, (unsigned int)ga,
                 (unsigned int)((ga >> 32) & 0x1FFFFFFu) | (2u << 30) };
    // D# group1: data_size=8B (code 3); tensor_dim0=2048, tensor_dim1=1;
    //            tile_dim0=2048, tile_dim1=1; tensor_dim0_stride=2048
    i32x8 g1 = { (int)(3u << 16),
                 (int)(2048u << 16),
                 (int)(1u << 16),
                 (int)(2048u << 16),
                 1,
                 2048,
                 0, 0 };
    i32x4 z4 = {0, 0, 0, 0};
#if __clang_major__ >= 23
    i32x8 z8 = {0, 0, 0, 0, 0, 0, 0, 0};
    __builtin_amdgcn_tensor_load_to_lds(g0, g1, z4, z4, z8, 0);
#else
    __builtin_amdgcn_tensor_load_to_lds(g0, g1, z4, z4, 0);
#endif
    __builtin_amdgcn_s_wait_tensorcnt(0);
  }
#else
  {
    const uint4* src4 = (const uint4*)Ai;
    uint4* dst4 = (uint4*)&xa[0];
    for (int i = tid; i < 16 * EDIM * 2 / 16; i += 256) dst4[i] = src4[i];
  }
#endif
  __syncthreads();

  // out2 = xa @ Wo + bo ; wave w -> output cols [w*32, w*32+32)
  {
    int n0 = w * 32;
    v8f c0 = zero8(), c1 = zero8();
    for (int kc = 0; kc < 16; ++kc) {
      v16bf a = load_a_frag(xa, EDIM, kc * 32);
      c0 = wmma_bf16(a, load_b_frag(WoT, EDIM, n0,      kc * 32), c0);
      c1 = wmma_bf16(a, load_b_frag(WoT, EDIM, n0 + 16, kc * 32), c1);
    }
#pragma unroll
    for (int r = 0; r < 8; ++r) {
      int M = r + ((lane >> 4) << 3);
      int N0 = n0 + (lane & 15), N1 = N0 + 16;
      float v0 = c0[r] + bo[N0];
      float v1 = c1[r] + bo[N1];
      out2f[M * DMODEL + N0] = v0; out2b[M * DMODEL + N0] = (__bf16)v0;
      out2f[M * DMODEL + N1] = v1; out2b[M * DMODEL + N1] = (__bf16)v1;
    }
  }
  __syncthreads();

  // h = relu(out2 @ W1 + b1) ; wave w -> DFF cols [w*256, w*256+256)
  {
    v16bf af[8];
#pragma unroll
    for (int kc = 0; kc < 8; ++kc) af[kc] = load_a_frag(out2b, DMODEL, kc * 32);
    for (int nt = 0; nt < 16; ++nt) {
      int n0 = w * 256 + nt * 16;
      v8f c = zero8();
#pragma unroll
      for (int kc = 0; kc < 8; ++kc)
        c = wmma_bf16(af[kc], load_b_frag(W1T, DMODEL, n0, kc * 32), c);
#pragma unroll
      for (int r = 0; r < 8; ++r) {
        int M = r + ((lane >> 4) << 3);
        int N = n0 + (lane & 15);
        hb[M * DFF + N] = (__bf16)fmaxf(c[r] + b1[N], 0.f);
      }
    }
  }
  __syncthreads();

  // y = h @ W2 + b2 + out2 ; wave w -> cols [w*32, w*32+32)
  {
    int n0 = w * 32;
    v8f c0 = zero8(), c1 = zero8();
    for (int kc = 0; kc < 64; ++kc) {
      v16bf a = load_a_frag(hb, DFF, kc * 32);
      c0 = wmma_bf16(a, load_b_frag(W2T, DFF, n0,      kc * 32), c0);
      c1 = wmma_bf16(a, load_b_frag(W2T, DFF, n0 + 16, kc * 32), c1);
    }
#pragma unroll
    for (int r = 0; r < 8; ++r) {
      int M = r + ((lane >> 4) << 3);
      int N0 = n0 + (lane & 15), N1 = N0 + 16;
      yf[M * DMODEL + N0] = c0[r] + b2[N0] + out2f[M * DMODEL + N0];
      yf[M * DMODEL + N1] = c1[r] + b2[N1] + out2f[M * DMODEL + N1];
    }
  }
  __syncthreads();

  // LayerNorm over the 256 features of each of the 16 rows
  {
    int row = tid >> 4, j = tid & 15;
    float s1 = 0.f, s2 = 0.f;
    for (int c = j * 16; c < j * 16 + 16; ++c) {
      float v = yf[row * DMODEL + c];
      s1 += v; s2 += v * v;
    }
    red1[row * 16 + j] = s1; red2[row * 16 + j] = s2;
    __syncthreads();
    if (tid < 16) {
      float a = 0.f, q = 0.f;
#pragma unroll
      for (int j2 = 0; j2 < 16; ++j2) { a += red1[tid * 16 + j2]; q += red2[tid * 16 + j2]; }
      float mu  = a * (1.f / DMODEL);
      float var = q * (1.f / DMODEL) - mu * mu;
      muA[tid] = mu;
      rsA[tid] = rsqrtf(var + 1e-5f);
    }
    __syncthreads();
    float mu = muA[row], rs = rsA[row];
    float* op = out + ((size_t)b * SEQ + s0 + row) * DMODEL;
    for (int c = j * 16; c < j * 16 + 16; ++c)
      op[c] = (yf[row * DMODEL + c] - mu) * rs * gamma[c] + beta[c];
  }
}

// ---------- Host launcher ----------
extern "C" void kernel_launch(void* const* d_in, const int* in_sizes, int n_in,
                              void* d_out, int out_size, void* d_ws, size_t ws_size,
                              hipStream_t stream) {
  (void)in_sizes; (void)n_in; (void)out_size; (void)ws_size;
  const float* plan  = (const float*)d_in[0];
  const float* knobs = (const float*)d_in[1];
  const float* Wmask = (const float*)d_in[2];
  const float* Wq    = (const float*)d_in[3];
  const float* Wk    = (const float*)d_in[4];
  const float* Wv    = (const float*)d_in[5];
  const float* Wo    = (const float*)d_in[6];
  const float* bo    = (const float*)d_in[7];
  const float* W1    = (const float*)d_in[8];
  const float* b1    = (const float*)d_in[9];
  const float* W2    = (const float*)d_in[10];
  const float* b2    = (const float*)d_in[11];
  const float* gamma = (const float*)d_in[12];
  const float* beta  = (const float*)d_in[13];

  char* ws = (char*)d_ws;
  size_t off = 0;
  auto take = [&](size_t bytes) {
    size_t r = (off + 255) & ~(size_t)255;
    off = r + bytes;
    return r;
  };
  float*  Wqp  = (float*)(ws + take((size_t)16 * EDIM * 4));
  float*  Wkp  = (float*)(ws + take((size_t)16 * EDIM * 4));
  float*  Wvp  = (float*)(ws + take((size_t)16 * EDIM * 4));
  __bf16* WoT  = (__bf16*)(ws + take((size_t)DMODEL * EDIM * 2));
  __bf16* W1T  = (__bf16*)(ws + take((size_t)DFF * DMODEL * 2));
  __bf16* W2T  = (__bf16*)(ws + take((size_t)DMODEL * DFF * 2));
  __bf16* Qb   = (__bf16*)(ws + take((size_t)NB * NH * SEQ * HD * 2));
  __bf16* Kb   = (__bf16*)(ws + take((size_t)NB * NH * SEQ * HD * 2));
  __bf16* Vt   = (__bf16*)(ws + take((size_t)NB * NH * HD * SEQ * 2));
  __bf16* Aout = (__bf16*)(ws + take((size_t)NB * SEQ * EDIM * 2));

  prep_wqkv<<<(3 * 16 * EDIM + 255) / 256, 256, 0, stream>>>(Wq, Wk, Wv, Wqp, Wkp, Wvp);
  transpose_bf16<<<(EDIM * DMODEL + 255) / 256, 256, 0, stream>>>(Wo, WoT, EDIM, DMODEL);
  transpose_bf16<<<(DMODEL * DFF + 255) / 256, 256, 0, stream>>>(W1, W1T, DMODEL, DFF);
  transpose_bf16<<<(DFF * DMODEL + 255) / 256, 256, 0, stream>>>(W2, W2T, DFF, DMODEL);
  qkv_proj<<<NB * SEQ, 256, 0, stream>>>(plan, knobs, Wqp, Wkp, Wvp, Qb, Kb, Vt);
  attn_kernel<<<NB * (SEQ / 16), 256, 0, stream>>>(Qb, Kb, Vt, Wmask, Aout);
  ffn_kernel<<<NB * (SEQ / 16), 256, 0, stream>>>(Aout, WoT, bo, W1T, b1, W2T, b2,
                                                  gamma, beta, (float*)d_out);
}
